// Attention_54193897341567
// MI455X (gfx1250) — compile-verified, matrix operand order checked by reading, not run on hardware
//
#include <hip/hip_runtime.h>
#include <hip/hip_bf16.h>
#include <math.h>
#include <stdint.h>

typedef __attribute__((ext_vector_type(16))) __bf16 v16bf;
typedef __attribute__((ext_vector_type(8)))  float  v8f;
typedef __attribute__((ext_vector_type(4)))  unsigned int v4u;
typedef __attribute__((ext_vector_type(8)))  int    v8i_t;
typedef __attribute__((ext_vector_type(4)))  int    v4i_t;

#define Bsz 2
#define Sq  2048
#define Dm  2048
#define Hh  16
#define HD  128

// ---------------------------------------------------------------------------
// WMMA helper: D = A(16x32 bf16) x B(32x16 bf16) + C(16x16 f32)
// ---------------------------------------------------------------------------
__device__ __forceinline__ v8f wmma_bf16(v16bf a, v16bf b, v8f c) {
  return __builtin_amdgcn_wmma_f32_16x16x32_bf16(
      /*neg_a=*/false, a, /*neg_b=*/false, b,
      /*c_mod=*/(short)0, c, /*reuse_a=*/false, /*reuse_b=*/false);
}

__device__ __forceinline__ v8f zero8() {
  v8f z = {0.f, 0.f, 0.f, 0.f, 0.f, 0.f, 0.f, 0.f};
  return z;
}

// K-pair index for element-pair p (0..7): ISA 05_wmma 16-bit A layout.
__device__ __forceinline__ int kpair(int p, int hi) {
  return (p & 3) + ((p >> 2) << 3) + (hi << 2);
}

// A/B fragment from a row-major bf16 tile: lane holds row/col (lane&15),
// elements run along K (the contiguous dim of src).
__device__ __forceinline__ v16bf load_frag_b16(const __bf16* src, int ld,
                                               int row0, int k0) {
  const int lane = threadIdx.x & 31;
  const int hi = lane >> 4;
  const __bf16* p = src + (size_t)(row0 + (lane & 15)) * ld + k0;
  v16bf f;
#pragma unroll
  for (int pi = 0; pi < 8; ++pi) {
    const int kp = kpair(pi, hi);
    f[2 * pi]     = p[2 * kp];
    f[2 * pi + 1] = p[2 * kp + 1];
  }
  return f;
}

// Same, but source is fp32 (LDS scores) converted to bf16 on load.
__device__ __forceinline__ v16bf load_frag_f32(const float* src, int ld,
                                               int row0, int k0) {
  const int lane = threadIdx.x & 31;
  const int hi = lane >> 4;
  const float* p = src + (size_t)(row0 + (lane & 15)) * ld + k0;
  v16bf f;
#pragma unroll
  for (int pi = 0; pi < 8; ++pi) {
    const int kp = kpair(pi, hi);
    f[2 * pi]     = (__bf16)p[2 * kp];
    f[2 * pi + 1] = (__bf16)p[2 * kp + 1];
  }
  return f;
}

// B fragment where K runs along ROWS of src (column access):
// B[k][n] = src[k*ld + n];  lane holds n = col0 + (lane&15).
__device__ __forceinline__ v16bf load_frag_col_b16(const __bf16* src, int ld,
                                                   int k0, int col0) {
  const int lane = threadIdx.x & 31;
  const int hi = lane >> 4;
  const __bf16* p = src + col0 + (lane & 15);
  v16bf f;
#pragma unroll
  for (int pi = 0; pi < 8; ++pi) {
    const int kp = kpair(pi, hi);
    f[2 * pi]     = p[(size_t)(k0 + 2 * kp)     * ld];
    f[2 * pi + 1] = p[(size_t)(k0 + 2 * kp + 1) * ld];
  }
  return f;
}

// ---------------------------------------------------------------------------
// TDM: 2D tile (tile_d0 elems x tile_d1 rows, bf16) global -> LDS, with
// 1-DWORD pad per 256B of LDS row (pad_interval=5, pad_amount=0).
// Destination LDS row stride = tile_d0 + 2 bf16 elements.
// ---------------------------------------------------------------------------
__device__ __forceinline__ void tdm_load_tile_bf16(unsigned int lds_addr,
                                                   const __bf16* gsrc,
                                                   int tile_d0, int tile_d1,
                                                   int row_stride_elems) {
  const unsigned long long ga = (unsigned long long)(uintptr_t)gsrc;
  v4u g0;
  g0[0] = 1u;                                   // count=1, no gather
  g0[1] = lds_addr;                             // lds_addr (bytes)
  g0[2] = (unsigned int)ga;                     // global_addr[31:0]
  g0[3] = ((unsigned int)(ga >> 32) & 0x01FFFFFFu) | (2u << 30);  // type=2
  v8i_t g1;
  g1[0] = (1 << 16)        // data_size = 2 bytes
        | (1 << 20)        // pad_enable
        | (5 << 22);       // pad_interval: every 64 DWORDs (256B)
                           // pad_amount = 0 -> 1 DWORD
  g1[1] = (tile_d0 & 0xFFFF) << 16;             // tensor_dim0[15:0]
  g1[2] = (tile_d1 & 0xFFFF) << 16;             // d0 hi=0 | tensor_dim1[15:0]
  g1[3] = (tile_d0 & 0xFFFF) << 16;             // d1 hi=0 | tile_dim0
  g1[4] = (tile_d1 & 0xFFFF);                   // tile_dim1 | tile_dim2=0
  g1[5] = row_stride_elems;                     // tensor_dim0_stride lo
  g1[6] = 0;
  g1[7] = 0;
  v4i_t z4 = {0, 0, 0, 0};
#if defined(__clang_major__) && (__clang_major__ >= 23)
  v8i_t z8 = {0, 0, 0, 0, 0, 0, 0, 0};
  __builtin_amdgcn_tensor_load_to_lds(g0, g1, z4, z4, z8, 0);
#else
  __builtin_amdgcn_tensor_load_to_lds(g0, g1, z4, z4, 0);
#endif
}

// 4 fp32 -> 4 bf16 packed in 8 bytes
union Pk4 {
  __bf16 e[4];
  unsigned long long u64;
};

// ---------------------------------------------------------------------------
// Kernel 1: Y = x @ W^T  (bf16 out).  grid = (N/128, M/128, 3) picks Wq/Wk/Wv
// block = 256 (8 waves); wave tile 32x64 (2x4 WMMA accumulators)
// ---------------------------------------------------------------------------
#define LDK 36   // row pitch 72B: 8B-aligned 4-elem groups, bank-staggered
__global__ void gemm_qkv(const float* __restrict__ x,
                         const float* __restrict__ Wq,
                         const float* __restrict__ Wk,
                         const float* __restrict__ Wv,
                         __bf16* __restrict__ Q, __bf16* __restrict__ K,
                         __bf16* __restrict__ V) {
  const int which = blockIdx.z;
  const float* W = (which == 0) ? Wq : (which == 1) ? Wk : Wv;
  __bf16* Y = (which == 0) ? Q : (which == 1) ? K : V;

  __shared__ __bf16 as[128 * LDK];
  __shared__ __bf16 bs[128 * LDK];

  const int m0 = blockIdx.y * 128, n0 = blockIdx.x * 128;
  const int tid = threadIdx.x, wave = tid >> 5;
  const int wm = wave & 3, wn = wave >> 2;  // 4x2 wave grid

  v8f acc[2][4];
#pragma unroll
  for (int i = 0; i < 2; ++i)
#pragma unroll
    for (int j = 0; j < 4; ++j) acc[i][j] = zero8();

  for (int k0 = 0; k0 < Dm; k0 += 32) {
    __syncthreads();
    // stage 128x32 fp32 -> bf16: float4 loads, 8B LDS stores
#pragma unroll
    for (int f = tid; f < 1024; f += 256) {
      const int r = f >> 3, c4 = (f & 7) * 4;
      const float4 xv = *(const float4*)&x[(size_t)(m0 + r) * Dm + k0 + c4];
      const float4 wv = *(const float4*)&W[(size_t)(n0 + r) * Dm + k0 + c4];
      Pk4 pa, pb;
      pa.e[0] = (__bf16)xv.x; pa.e[1] = (__bf16)xv.y;
      pa.e[2] = (__bf16)xv.z; pa.e[3] = (__bf16)xv.w;
      pb.e[0] = (__bf16)wv.x; pb.e[1] = (__bf16)wv.y;
      pb.e[2] = (__bf16)wv.z; pb.e[3] = (__bf16)wv.w;
      *(unsigned long long*)&as[r * LDK + c4] = pa.u64;
      *(unsigned long long*)&bs[r * LDK + c4] = pb.u64;
    }
    // prefetch next K-tile (one cacheline per thread) -> global_prefetch_b8
    if (k0 + 32 < Dm) {
      const int pr = tid >> 1;
      const float* pb = (tid & 1)
                            ? &W[(size_t)(n0 + pr) * Dm + k0 + 32]
                            : &x[(size_t)(m0 + pr) * Dm + k0 + 32];
      __builtin_prefetch(pb, 0, 0);
    }
    __syncthreads();

    v16bf afr[2];
#pragma unroll
    for (int mt = 0; mt < 2; ++mt)
      afr[mt] = load_frag_b16(as, LDK, wm * 32 + mt * 16, 0);
#pragma unroll
    for (int nt = 0; nt < 4; ++nt) {
      v16bf bfr = load_frag_b16(bs, LDK, wn * 64 + nt * 16, 0);
#pragma unroll
      for (int mt = 0; mt < 2; ++mt)
        acc[mt][nt] = wmma_bf16(afr[mt], bfr, acc[mt][nt]);
    }
  }

  const int lane = tid & 31, hi = lane >> 4, l16 = lane & 15;
#pragma unroll
  for (int mt = 0; mt < 2; ++mt)
#pragma unroll
    for (int nt = 0; nt < 4; ++nt)
#pragma unroll
      for (int r = 0; r < 8; ++r) {
        const int row = m0 + wm * 32 + mt * 16 + r + hi * 8;
        const int col = n0 + wn * 64 + nt * 16 + l16;
        Y[(size_t)row * Dm + col] = (__bf16)acc[mt][nt][r];
      }
}

// ---------------------------------------------------------------------------
// Kernel 2: interleaved-pair RoPE on Q and K (bf16, in place, 16B vectors)
// Each thread rotates 4 adjacent pairs (8 bf16) of one row.
// ---------------------------------------------------------------------------
__global__ void rope_kernel(__bf16* __restrict__ Q, __bf16* __restrict__ K) {
  const size_t g = (size_t)blockIdx.x * blockDim.x + threadIdx.x;
  const size_t ngrp = (size_t)Bsz * Sq * (Dm / 8);
  if (g >= ngrp) return;
  const size_t row = g / (Dm / 8);
  const int col0 = (int)(g % (Dm / 8)) * 8;
  const int s = (int)(row % Sq);       // sequence position

  float cs[4], sn[4];
#pragma unroll
  for (int p = 0; p < 4; ++p) {
    const int d = (col0 + 2 * p) & (HD - 1);   // dim within head
    const float theta = __expf(-((float)d / (float)HD) * 9.210340371976184f);
    __sincosf((float)s * theta, &sn[p], &cs[p]);
  }

  union U8 { __bf16 e[8]; uint4 u; };
  const size_t o = row * Dm + col0;
  {
    U8 q; q.u = *(const uint4*)&Q[o];
#pragma unroll
    for (int p = 0; p < 4; ++p) {
      const float x0 = (float)q.e[2 * p], x1 = (float)q.e[2 * p + 1];
      q.e[2 * p]     = (__bf16)(x0 * cs[p] - x1 * sn[p]);
      q.e[2 * p + 1] = (__bf16)(x0 * sn[p] + x1 * cs[p]);
    }
    *(uint4*)&Q[o] = q.u;
  }
  {
    U8 k; k.u = *(const uint4*)&K[o];
#pragma unroll
    for (int p = 0; p < 4; ++p) {
      const float x0 = (float)k.e[2 * p], x1 = (float)k.e[2 * p + 1];
      k.e[2 * p]     = (__bf16)(x0 * cs[p] - x1 * sn[p]);
      k.e[2 * p + 1] = (__bf16)(x0 * sn[p] + x1 * cs[p]);
    }
    *(uint4*)&K[o] = k.u;
  }
}

// ---------------------------------------------------------------------------
// Kernel 3: fused attention per (b, h, 32-row tile).
//   TDM Q tile + TDM-chunked K (128 rows) -> scores (WMMA, causal) -> softmax
//   -> NT-stream A (write-once) -> TDM double-buffered V -> P@V -> bf16 O
// grid = (S/32, H, B), block = 256 (8 waves)
// ---------------------------------------------------------------------------
#define SLD 2052
#define QLD (HD + 2)   // 128 elems + 1 DWORD TDM pad per row = 130 bf16
__global__ void attn_kernel(const __bf16* __restrict__ Q,
                            const __bf16* __restrict__ Kb,
                            const __bf16* __restrict__ Vb,
                            float* __restrict__ A_out,
                            __bf16* __restrict__ O) {
  __shared__ __bf16 qtile[32 * QLD];     // Q tile (TDM, padded rows)
  __shared__ __bf16 stg[128 * QLD];      // K chunk / double-buffered V chunks
  extern __shared__ float smem[];
  float* sc   = smem;                 // [32][SLD] scores -> probabilities
  float* red  = smem + 32 * SLD;      // [32][8]
  float* rmax = red + 32 * 8;         // [32]
  float* rsum = rmax + 32;            // [32]

  const int i0 = blockIdx.x * 32;
  const int h = blockIdx.y, b = blockIdx.z;
  const int tid = threadIdx.x, wave = tid >> 5;
  const int lane = tid & 31, hi = lane >> 4, l16 = lane & 15;

  const __bf16* Qh = Q  + (size_t)b * Sq * Dm + h * HD;
  const __bf16* Kh = Kb + (size_t)b * Sq * Dm + h * HD;
  const __bf16* Vh = Vb + (size_t)b * Sq * Dm + h * HD;

  const int jlim = i0 + 32;           // exclusive causal column limit
  const int nJT = jlim >> 4;          // 16-wide j-tiles to compute
  const float scale = 0.08838834764831845f;  // 1/sqrt(128)

  const unsigned int qtile_a = (unsigned int)(uintptr_t)qtile;
  const unsigned int stg_a   = (unsigned int)(uintptr_t)stg;

  // ---- TDM: DMA the 32x128 bf16 Q tile into LDS ----
  if (tid < 32) {
    tdm_load_tile_bf16(qtile_a, Qh + (size_t)i0 * Dm, HD, 32, Dm);
    __builtin_amdgcn_s_wait_tensorcnt(0);
  }
  __syncthreads();

  // Q fragments: loaded once from LDS, reused across all j-tiles
  v16bf qa[2][4];
#pragma unroll
  for (int mt = 0; mt < 2; ++mt)
#pragma unroll
    for (int kk = 0; kk < 4; ++kk)
      qa[mt][kk] = load_frag_b16(qtile, QLD, mt * 16, kk * 32);

  // ---- scores: TDM-stage K in 128-row chunks; 1 j-tile per wave ----
  for (int j0 = 0; j0 < jlim; j0 += 128) {
    if (tid < 32) {
      tdm_load_tile_bf16(stg_a, Kh + (size_t)j0 * Dm, HD, 128, Dm);
      __builtin_amdgcn_s_wait_tensorcnt(0);
    }
    __syncthreads();

    const int jt = (j0 >> 4) + wave;   // this wave's 16-wide j-tile
    if (jt < nJT) {
      v8f acc0 = zero8(), acc1 = zero8();
#pragma unroll
      for (int kk = 0; kk < 4; ++kk) {
        v16bf bk = load_frag_b16(stg, QLD, wave * 16, kk * 32);  // lane = j
        acc0 = wmma_bf16(qa[0][kk], bk, acc0);
        acc1 = wmma_bf16(qa[1][kk], bk, acc1);
      }
#pragma unroll
      for (int mt = 0; mt < 2; ++mt) {
        v8f& a = mt ? acc1 : acc0;
#pragma unroll
        for (int r = 0; r < 8; ++r) {
          const int il = mt * 16 + r + hi * 8;
          const int ig = i0 + il;
          const int jg = jt * 16 + l16;
          float v = a[r] * scale;
          if (jg > ig) v = -__builtin_inff();
          sc[il * SLD + jg] = v;
        }
      }
    }
    __syncthreads();  // chunk consumed before next TDM overwrite
  }

  // Kick off V chunk 0 DMA; it overlaps the softmax below.
  const int nChR = (jlim + 63) >> 6;  // 64-row V chunks (+ 32-row tail case)
  if (tid < 32)
    tdm_load_tile_bf16(stg_a, Vh, HD, (jlim >= 64) ? 64 : 32, Dm);

  // ---- softmax: 8 threads per row ----
  const int rid = tid >> 3, sub = tid & 7;
  float m = -__builtin_inff();
  for (int j = sub; j < jlim; j += 8) m = fmaxf(m, sc[rid * SLD + j]);
  red[rid * 8 + sub] = m;
  __syncthreads();
  if (sub == 0) {
    float mm = red[rid * 8];
#pragma unroll
    for (int t = 1; t < 8; ++t) mm = fmaxf(mm, red[rid * 8 + t]);
    rmax[rid] = mm;
  }
  __syncthreads();
  const float mm = rmax[rid];
  float ls = 0.f;
  for (int j = sub; j < jlim; j += 8) ls += __expf(sc[rid * SLD + j] - mm);
  red[rid * 8 + sub] = ls;
  __syncthreads();
  if (sub == 0) {
    float s = 0.f;
#pragma unroll
    for (int t = 0; t < 8; ++t) s += red[rid * 8 + t];
    rsum[rid] = s;
  }
  __syncthreads();
  const float inv = 1.0f / rsum[rid];

  // ---- normalize in LDS + NT-stream A out (write-once, bypass caches) ----
  const size_t abase = ((size_t)(b * Hh + h)) * Sq * Sq;
  float* Ar = A_out + abase + (size_t)(i0 + rid) * Sq;
  for (int j = sub; j < jlim; j += 8) {
    const float p = __expf(sc[rid * SLD + j] - mm) * inv;
    sc[rid * SLD + j] = p;
    __builtin_nontemporal_store(p, &Ar[j]);
  }
  for (int j = jlim + sub; j < Sq; j += 8)
    __builtin_nontemporal_store(0.f, &Ar[j]);  // masked region
  __syncthreads();

  // ---- O = P @ V : double-buffered 64-row V chunks; wave w owns d-tile w ----
  const int dt = wave * 16;
  v8f oa0 = zero8(), oa1 = zero8();
  for (int c = 0; c < nChR; ++c) {
    const int j0 = c * 64;
    if (tid < 32) {
      if (c + 1 < nChR) {   // prefetch next chunk into the other half
        const int jn = (c + 1) * 64;
        const int rows = (jlim - jn >= 64) ? 64 : 32;
        tdm_load_tile_bf16(stg_a + (unsigned)(((c + 1) & 1) * 64 * QLD * 2),
                           Vh + (size_t)jn * Dm, HD, rows, Dm);
        __builtin_amdgcn_s_wait_tensorcnt(1);  // chunk c landed (in-order)
      } else {
        __builtin_amdgcn_s_wait_tensorcnt(0);
      }
    }
    __syncthreads();

    const __bf16* vbuf = stg + (c & 1) * 64 * QLD;
    const int nj = ((jlim - j0) >= 64) ? 2 : 1;
#pragma unroll
    for (int jj = 0; jj < 2; ++jj) {
      if (jj < nj) {
        v16bf vb = load_frag_col_b16(vbuf, QLD, jj * 32, dt);
        v16bf p0 = load_frag_f32(sc, SLD, 0,  j0 + jj * 32);
        v16bf p1 = load_frag_f32(sc, SLD, 16, j0 + jj * 32);
        oa0 = wmma_bf16(p0, vb, oa0);
        oa1 = wmma_bf16(p1, vb, oa1);
      }
    }
    __syncthreads();  // buffer free for the chunk issued next iteration
  }

#pragma unroll
  for (int mt = 0; mt < 2; ++mt) {
    v8f& a = mt ? oa1 : oa0;
#pragma unroll
    for (int r = 0; r < 8; ++r) {
      const int row = b * Sq + i0 + mt * 16 + r + hi * 8;
      const int col = h * HD + dt + l16;
      O[(size_t)row * Dm + col] = (__bf16)a[r];
    }
  }
}

// ---------------------------------------------------------------------------
// Kernel 4: attention = O @ Wo^T + bo  (fp32 out)
// ---------------------------------------------------------------------------
__global__ void gemm_out(const __bf16* __restrict__ Oin,
                         const float* __restrict__ Wo,
                         const float* __restrict__ bo,
                         float* __restrict__ out) {
  __shared__ __bf16 as[128 * LDK];
  __shared__ __bf16 bs[128 * LDK];

  const int m0 = blockIdx.y * 128, n0 = blockIdx.x * 128;
  const int tid = threadIdx.x, wave = tid >> 5;
  const int wm = wave & 3, wn = wave >> 2;

  v8f acc[2][4];
#pragma unroll
  for (int i = 0; i < 2; ++i)
#pragma unroll
    for (int j = 0; j < 4; ++j) acc[i][j] = zero8();

  for (int k0 = 0; k0 < Dm; k0 += 32) {
    __syncthreads();
    // O tile: bf16 source, 16B loads + two 8B LDS stores
#pragma unroll
    for (int f = tid; f < 512; f += 256) {
      const int r = f >> 2, c8 = (f & 3) * 8;
      const uint4 ov = *(const uint4*)&Oin[(size_t)(m0 + r) * Dm + k0 + c8];
      *(unsigned long long*)&as[r * LDK + c8] =
          ((unsigned long long)ov.y << 32) | ov.x;
      *(unsigned long long*)&as[r * LDK + c8 + 4] =
          ((unsigned long long)ov.w << 32) | ov.z;
    }
    // Wo tile: fp32 -> bf16, float4 loads, 8B LDS stores
#pragma unroll
    for (int f = tid; f < 1024; f += 256) {
      const int r = f >> 3, c4 = (f & 7) * 4;
      const float4 wv = *(const float4*)&Wo[(size_t)(n0 + r) * Dm + k0 + c4];
      Pk4 pb;
      pb.e[0] = (__bf16)wv.x; pb.e[1] = (__bf16)wv.y;
      pb.e[2] = (__bf16)wv.z; pb.e[3] = (__bf16)wv.w;
      *(unsigned long long*)&bs[r * LDK + c4] = pb.u64;
    }
    if (k0 + 32 < Dm) {
      const int pr = tid >> 1;
      const void* pb = (tid & 1)
                           ? (const void*)&Wo[(size_t)(n0 + pr) * Dm + k0 + 32]
                           : (const void*)&Oin[(size_t)(m0 + pr) * Dm + k0 + 32];
      __builtin_prefetch(pb, 0, 0);
    }
    __syncthreads();

    v16bf afr[2];
#pragma unroll
    for (int mt = 0; mt < 2; ++mt)
      afr[mt] = load_frag_b16(as, LDK, wm * 32 + mt * 16, 0);
#pragma unroll
    for (int nt = 0; nt < 4; ++nt) {
      v16bf bfr = load_frag_b16(bs, LDK, wn * 64 + nt * 16, 0);
#pragma unroll
      for (int mt = 0; mt < 2; ++mt)
        acc[mt][nt] = wmma_bf16(afr[mt], bfr, acc[mt][nt]);
    }
  }

  const int lane = tid & 31, hi = lane >> 4, l16 = lane & 15;
#pragma unroll
  for (int mt = 0; mt < 2; ++mt)
#pragma unroll
    for (int nt = 0; nt < 4; ++nt)
#pragma unroll
      for (int r = 0; r < 8; ++r) {
        const int row = m0 + wm * 32 + mt * 16 + r + hi * 8;
        const int col = n0 + wn * 64 + nt * 16 + l16;
        out[(size_t)row * Dm + col] = acc[mt][nt][r] + bo[col];
      }
}

// ---------------------------------------------------------------------------
extern "C" void kernel_launch(void* const* d_in, const int* in_sizes, int n_in,
                              void* d_out, int out_size, void* d_ws,
                              size_t ws_size, hipStream_t stream) {
  const float* x  = (const float*)d_in[0];
  const float* Wq = (const float*)d_in[1];
  const float* Wk = (const float*)d_in[2];
  const float* Wv = (const float*)d_in[3];
  const float* Wo = (const float*)d_in[4];
  const float* bo = (const float*)d_in[5];

  float* attn_out = (float*)d_out;                       // [B,S,D]
  float* A_out = attn_out + (size_t)Bsz * Sq * Dm;       // [B,H,S,S]

  const size_t nElem = (size_t)Bsz * Sq * Dm;            // 8,388,608
  char* wsb = (char*)d_ws;
  __bf16* Qb = (__bf16*)(wsb);
  __bf16* Kb = (__bf16*)(wsb + nElem * 2);
  __bf16* Vb = (__bf16*)(wsb + nElem * 4);
  __bf16* Ob = (__bf16*)(wsb + nElem * 6);

  // 1. QKV projections -> bf16
  gemm_qkv<<<dim3(Dm / 128, (Bsz * Sq) / 128, 3), 256, 0, stream>>>(
      x, Wq, Wk, Wv, Qb, Kb, Vb);

  // 2. RoPE on Q, K (vectorized: 4 pairs / thread)
  {
    const size_t ngrp = nElem / 8;
    const int blocks = (int)((ngrp + 255) / 256);
    rope_kernel<<<blocks, 256, 0, stream>>>(Qb, Kb);
  }

  // 3. fused attention: TDM-staged Q/K/V -> scores -> softmax -> A + P@V
  {
    const size_t shmem = (size_t)(32 * SLD + 32 * 8 + 64) * sizeof(float);
    attn_kernel<<<dim3(Sq / 32, Hh, Bsz), 256, shmem, stream>>>(
        Qb, Kb, Vb, A_out, Ob);
  }

  // 4. output projection + bias -> fp32
  gemm_out<<<dim3(Dm / 128, (Bsz * Sq) / 128, 1), 256, 0, stream>>>(
      Ob, Wo, bo, attn_out);
}
